// LstmCellWithProjection_5669356834275
// MI455X (gfx1250) — matrix operand order repaired and checked by reading, not compiled, not run on hardware
//
#include <hip/hip_runtime.h>
#include <hip/hip_bf16.h>
#include <stdint.h>

// Problem dims (fixed by reference)
#define B_   64
#define T_   512
#define I_   512
#define H_   512
#define C_   4096
#define G_   (4 * C_)      // 16384 gate rows
#define KCAT (I_ + H_)     // 1024 fused K for [x|h] GEMM

typedef __attribute__((ext_vector_type(16))) __bf16 v16bf;
typedef __attribute__((ext_vector_type(8)))  float  v8f;
typedef uint64_t u64;

// d_out layout: output (B,T,H) | final_h (B,H) | final_c (B,C)
#define OUT_OFF_H ((size_t)B_ * T_ * H_)
#define OUT_OFF_C (OUT_OFF_H + (size_t)B_ * H_)

// workspace layout (bytes)
#define WS_XBF   ((size_t)0)
#define WS_WCAT  (WS_XBF   + (size_t)B_ * T_ * I_ * 2)   // x in bf16
#define WS_WPROJ (WS_WCAT  + (size_t)G_ * KCAT * 2)      // [Wi|Ws] bf16
#define WS_HBF   (WS_WPROJ + (size_t)H_ * C_ * 2)        // W_proj bf16
#define WS_ACT   (WS_HBF   + (size_t)B_ * H_ * 2)        // h bf16

__device__ __forceinline__ float sigmoid_f(float x) {
    return 1.0f / (1.0f + __expf(-x));
}
__device__ __forceinline__ float tanh_f(float x) {
    float e = __expf(2.0f * x);
    return (e - 1.0f) / (e + 1.0f);
}
__device__ __forceinline__ float clip3(float x) {
    return fminf(fmaxf(x, -3.0f), 3.0f);
}

// ---------------- prep kernels ----------------
__global__ void k_f32_to_bf16(const float* __restrict__ src, __bf16* __restrict__ dst, size_t n) {
    size_t i = (size_t)blockIdx.x * blockDim.x + threadIdx.x;
    size_t stride = (size_t)gridDim.x * blockDim.x;
    for (; i < n; i += stride) dst[i] = (__bf16)src[i];
}

// Build concatenated weight (G rows x KCAT cols), row-major over K, bf16.
__global__ void k_build_wcat(const float* __restrict__ wi, const float* __restrict__ ws,
                             __bf16* __restrict__ wcat) {
    size_t n = (size_t)G_ * KCAT;
    size_t i = (size_t)blockIdx.x * blockDim.x + threadIdx.x;
    size_t stride = (size_t)gridDim.x * blockDim.x;
    for (; i < n; i += stride) {
        size_t g = i / KCAT, k = i % KCAT;
        float v = (k < I_) ? wi[g * I_ + k] : ws[g * H_ + (k - I_)];
        wcat[i] = (__bf16)v;
    }
}

__global__ void k_init(float* __restrict__ h, float* __restrict__ c, __bf16* __restrict__ hbf) {
    int i = blockIdx.x * blockDim.x + threadIdx.x;
    int stride = gridDim.x * blockDim.x;
    for (int j = i; j < B_ * H_; j += stride) { h[j] = 0.0f; hbf[j] = (__bf16)0.0f; }
    for (int j = i; j < B_ * C_; j += stride) c[j] = 0.0f;
}

// ---------------- fused gates GEMM + LSTM cell ----------------
// grid = 256 blocks x 128 threads; block: mtile = blk&3, 4 waves -> 4 n-tiles.
// Each wave computes the 16x16 cell tile at (m0,n0) by accumulating the four
// gate tiles (rows g*C + n0..n0+15 of wcat) over K=1024 with bf16 WMMA, then
// runs the LSTM cell epilogue in registers.
__global__ __launch_bounds__(128)
void k_gates_cell(const __bf16* __restrict__ xbf, const __bf16* __restrict__ hbf,
                  const __bf16* __restrict__ wcat, const float* __restrict__ b_state,
                  const int* __restrict__ lengths, float* __restrict__ c_state,
                  __bf16* __restrict__ act, int t) {
    __shared__ __bf16 tile[16 * KCAT];  // 32 KB A strip: 16 batch rows x [x_t | h]

    const int tid  = threadIdx.x;
    const int wave = tid >> 5, lane = tid & 31;
    const int blk  = blockIdx.x;
    const int m0   = (blk & 3) * 16;
    const int n0   = ((blk >> 2) * 4 + wave) * 16;

    // --- async-stage A strip into LDS: 2 KB/row * 16 rows, 16B per copy ---
    for (int i = 0; i < 8; ++i) {
        int c = tid + i * 128;            // 1024 chunks per half
        int r = c >> 6;                   // row 0..15
        int k = (c & 63) * 8;             // 0..504, 8 bf16 = 16B
        const __bf16* gx = xbf + ((size_t)(m0 + r) * T_ + t) * I_ + k;
        unsigned lx = (unsigned)(uintptr_t)(tile + r * KCAT + k);
        asm volatile("global_load_async_to_lds_b128 %0, %1, off"
                     :: "v"(lx), "v"((u64)(uintptr_t)gx) : "memory");
        const __bf16* gh = hbf + (size_t)(m0 + r) * H_ + k;
        unsigned lh = (unsigned)(uintptr_t)(tile + r * KCAT + I_ + k);
        asm volatile("global_load_async_to_lds_b128 %0, %1, off"
                     :: "v"(lh), "v"((u64)(uintptr_t)gh) : "memory");
    }
    asm volatile("s_wait_asynccnt 0x0" ::: "memory");
    __syncthreads();

    v8f acc[4];
    #pragma unroll
    for (int g = 0; g < 4; ++g) acc[g] = (v8f){0.f,0.f,0.f,0.f,0.f,0.f,0.f,0.f};

    const int nrow = lane & 15;          // A: M-row / B: N-col within tile
    const int hi   = (lane >> 4) & 1;    // lane half selects K sub-chunk
    const __bf16* arow = tile + nrow * KCAT + hi * 8;

    for (int k0 = 0; k0 < KCAT; k0 += 32) {
        v16bf a;  // A 16x32: lane chunks at K {hi*8 .. +7} and {hi*8+16 .. +23}
        ((uint4*)&a)[0] = *(const uint4*)(arow + k0);
        ((uint4*)&a)[1] = *(const uint4*)(arow + k0 + 16);
        #pragma unroll
        for (int g = 0; g < 4; ++g) {
            // B 32x16: lane = col n0+nrow, K chunks at k0 + {hi*8 .. +7} and {hi*8+16 .. +23}
            const __bf16* wp = wcat + ((size_t)g * C_ + n0 + nrow) * KCAT + k0 + hi * 8;
            __builtin_prefetch(wp + 128, 0, 0);  // stream next weight chunk (L2)
            v16bf b;
            ((uint4*)&b)[0] = ((const uint4*)wp)[0];
            ((uint4*)&b)[1] = *(const uint4*)(wp + 16);
            acc[g] = __builtin_amdgcn_wmma_f32_16x16x32_bf16(
                false, a, false, b, (short)0, acc[g], false, false);
        }
    }

    // --- fused LSTM cell epilogue: acc[v] holds (row m0+v+hi*8, col n0+nrow) ---
    const int col = n0 + nrow;
    const float bi = b_state[0 * C_ + col];
    const float bf = b_state[1 * C_ + col];
    const float bm = b_state[2 * C_ + col];
    const float bo = b_state[3 * C_ + col];
    #pragma unroll
    for (int e = 0; e < 8; ++e) {
        int row = m0 + e + hi * 8;
        float ig = sigmoid_f(acc[0][e] + bi);
        float fg = sigmoid_f(acc[1][e] + bf);
        float mi = tanh_f(acc[2][e] + bm);
        float og = sigmoid_f(acc[3][e] + bo);
        float cold = c_state[(size_t)row * C_ + col];
        float mem  = clip3(ig * mi + fg * cold);
        bool active = t < lengths[row];
        c_state[(size_t)row * C_ + col] = active ? mem : cold;
        act[(size_t)row * C_ + col] = (__bf16)(og * tanh_f(mem));  // proj input
    }
}

// ---------------- projection GEMM + mask + h update ----------------
// out(64x512) = act(64x4096) @ W_proj^T ; grid = 32 blocks x 128 threads.
__global__ __launch_bounds__(128)
void k_proj(const __bf16* __restrict__ act, const __bf16* __restrict__ wproj,
            const int* __restrict__ lengths, float* __restrict__ out,
            float* __restrict__ h_state, __bf16* __restrict__ hbf, int t) {
    const int tid  = threadIdx.x;
    const int wave = tid >> 5, lane = tid & 31;
    const int blk  = blockIdx.x;
    const int m0   = (blk & 3) * 16;
    const int n0   = ((blk >> 2) * 4 + wave) * 16;
    const int nrow = lane & 15;
    const int hi   = (lane >> 4) & 1;

    v8f acc = (v8f){0.f,0.f,0.f,0.f,0.f,0.f,0.f,0.f};
    const __bf16* arow = act + (size_t)(m0 + nrow) * C_ + hi * 8;
    const __bf16* wrow = wproj + (size_t)(n0 + nrow) * C_ + hi * 8;

    for (int k0 = 0; k0 < C_; k0 += 32) {
        v16bf a;
        ((uint4*)&a)[0] = *(const uint4*)(arow + k0);
        ((uint4*)&a)[1] = *(const uint4*)(arow + k0 + 16);
        __builtin_prefetch(wrow + k0 + 256, 0, 0);
        v16bf b;
        ((uint4*)&b)[0] = ((const uint4*)(wrow + k0))[0];
        ((uint4*)&b)[1] = *(const uint4*)(wrow + k0 + 16);
        acc = __builtin_amdgcn_wmma_f32_16x16x32_bf16(
            false, a, false, b, (short)0, acc, false, false);
    }

    const int col = n0 + nrow;
    #pragma unroll
    for (int e = 0; e < 8; ++e) {
        int row = m0 + e + hi * 8;
        float v = clip3(acc[e]);
        bool active = t < lengths[row];
        out[((size_t)row * T_ + t) * H_ + col] = active ? v : 0.0f;
        if (active) {
            h_state[(size_t)row * H_ + col] = v;           // final_h region
            hbf[(size_t)row * H_ + col]     = (__bf16)v;   // next-step GEMM input
        }
    }
}

// ---------------- launch ----------------
extern "C" void kernel_launch(void* const* d_in, const int* in_sizes, int n_in,
                              void* d_out, int out_size, void* d_ws, size_t ws_size,
                              hipStream_t stream) {
    const float* x       = (const float*)d_in[0];
    const int*   lengths = (const int*)d_in[1];
    const float* Wi      = (const float*)d_in[2];
    const float* Ws      = (const float*)d_in[3];
    const float* bs      = (const float*)d_in[4];
    const float* Wp      = (const float*)d_in[5];

    float* out     = (float*)d_out;
    float* h_state = out + OUT_OFF_H;
    float* c_state = out + OUT_OFF_C;

    char* ws = (char*)d_ws;
    __bf16* xbf  = (__bf16*)(ws + WS_XBF);
    __bf16* wcat = (__bf16*)(ws + WS_WCAT);
    __bf16* wpb  = (__bf16*)(ws + WS_WPROJ);
    __bf16* hbf  = (__bf16*)(ws + WS_HBF);
    __bf16* act  = (__bf16*)(ws + WS_ACT);

    // One-time (per call) conversions; weights stay L2-resident thereafter.
    k_f32_to_bf16<<<4096, 256, 0, stream>>>(x, xbf, (size_t)B_ * T_ * I_);
    k_build_wcat<<<4096, 256, 0, stream>>>(Wi, Ws, wcat);
    k_f32_to_bf16<<<1024, 256, 0, stream>>>(Wp, wpb, (size_t)H_ * C_);
    k_init<<<512, 256, 0, stream>>>(h_state, c_state, hbf);

    for (int t = 0; t < T_; ++t) {
        k_gates_cell<<<256, 128, 0, stream>>>(xbf, hbf, wcat, bs, lengths, c_state, act, t);
        k_proj<<<32, 128, 0, stream>>>(act, wpb, lengths, out, h_state, hbf, t);
    }
}
